// Allegro_MACE_Module_88811333747234
// MI455X (gfx1250) — compile-verified, matrix operand order checked by reading, not run on hardware
//
#include <hip/hip_runtime.h>
#include <hip/hip_bf16.h>

// ---------------------------------------------------------------------------
// Allegro/MACE fused edge-message module for MI455X (gfx1250, wave32, WMMA).
// Precision strategy: all dense GEMMs run as V_WMMA_F32_16X16X32_F16
// (f16 operands, f32 accumulate); scatters via global_atomic_add_f32 into
// L2-resident node buffers; small 32x32 channel mixes as VALU.
// Epilogues use v_rcp_f32/v_sqrt_f32 hardware approximations to keep the
// VALU tail short so it co-issues with the WMMA pipe.
// ---------------------------------------------------------------------------

typedef __attribute__((ext_vector_type(16))) _Float16 v16h;
typedef __attribute__((ext_vector_type(8)))  float    v8f;

#define E_TOT   320000
#define NN_TOT  10000
#define C_OLD   0.8944271909999159f   // 1/sqrt(sigmoid(0)^2+1)
#define C_NEW   0.4472135954999579f

__device__ __constant__ int LOFF_C[3] = {0, 1, 4};
__device__ __constant__ int LDIM_C[3] = {1, 3, 5};

// silu via hardware v_rcp_f32 (avoids IEEE div expansion in the epilogue)
__device__ __forceinline__ float silu_f(float x) {
  return x * __builtin_amdgcn_rcpf(1.f + __expf(-x));
}

__device__ __forceinline__ void atomic_add_g(float* p, float v) {
  __hip_atomic_fetch_add(p, v, __ATOMIC_RELAXED, __HIP_MEMORY_SCOPE_AGENT);
}

// D(16x16,f32) += A(16xK from LDS, row-major f16) * B(Kx16 from global,
// stored column-major-transposed f16: Bt[n][kpad]).  K = 32*KC.
// A-fragment: lane L holds row (L&15), halves K = (L>>4)*16 + i  (contiguous k order, same as B)
// B-fragment: lane L holds col (L&15), halves K = (L>>4)*16 + i
template <int KC>
__device__ __forceinline__ v8f wmma_rowtile(const _Float16* __restrict__ A, int lda,
                                            const _Float16* __restrict__ Bt, int ldb,
                                            int col0, int lane, v8f acc) {
  const int idx = lane & 15;
  const int kb  = (lane >> 4) * 16;
#pragma unroll
  for (int c = 0; c < KC; ++c) {
    v16h a = *(const v16h*)(A + idx * lda + c * 32 + kb);
    v16h b = *(const v16h*)(Bt + (size_t)(col0 + idx) * ldb + c * 32 + kb);
    acc = __builtin_amdgcn_wmma_f32_16x16x32_f16(false, a, false, b,
                                                 (short)0, acc, false, false);
  }
  return acc;
}

// store D tile with silu -> f16 LDS buffer [16][ldd]
__device__ __forceinline__ void store_silu_h(v8f acc, _Float16* dst, int ldd,
                                             int col0, int lane) {
  const int n = lane & 15, hi = lane >> 4;
#pragma unroll
  for (int r = 0; r < 8; ++r) {
    const int m = r + 8 * hi;
    dst[m * ldd + col0 + n] = (_Float16)silu_f(acc[r]);
  }
}

// ---------------------------------------------------------------------------
// Weight prep: f32 [K][N] row-major  ->  f16 transposed [N][Kpad] (zero pad)
// ---------------------------------------------------------------------------
__global__ void allegro_prep_weight(const float* __restrict__ W,
                                    _Float16* __restrict__ Wt,
                                    int K, int N, int Kpad) {
  int total = N * Kpad;
  for (int i = blockIdx.x * blockDim.x + threadIdx.x; i < total;
       i += gridDim.x * blockDim.x) {
    int n = i / Kpad, k = i - n * Kpad;
    Wt[i] = (k < K) ? (_Float16)W[(size_t)k * N + n] : (_Float16)0.f;
  }
}

// ---------------------------------------------------------------------------
// Edge pass 1: latents = cutoff*silu(silu([ninv_c|ninv_n|einv]@W2a)@W2b)
//              w_all = latents@Wenv0 ; store w0, scatter env_e -> env0
// ---------------------------------------------------------------------------
__global__ void __launch_bounds__(128)
allegro_edge_pass1(const float* __restrict__ eattr, const float* __restrict__ elen,
                   const float* __restrict__ einv,  const float* __restrict__ ninv,
                   const int* __restrict__ eidx,
                   const _Float16* __restrict__ W2aT, const _Float16* __restrict__ W2bT,
                   const _Float16* __restrict__ Wenv0T,
                   float* __restrict__ latents, float* __restrict__ cutg,
                   float* __restrict__ w0g, float* __restrict__ env0) {
  __shared__ __align__(32) _Float16 sA[16 * 160];
  __shared__ __align__(32) _Float16 sB1[16 * 256];
  __shared__ __align__(32) _Float16 sB2[16 * 256];
  __shared__ float sWall[16 * 192];
  __shared__ float sSh[16 * 9];
  __shared__ float sCut[16];
  __shared__ int   sCtr[16];

  const int tid = threadIdx.x;
  const int lane = tid & 31, wv = tid >> 5;
  const size_t e0 = (size_t)blockIdx.x * 16;

  if (tid < 16) {
    size_t e = e0 + tid;
    sCtr[tid] = eidx[e];
    float x = elen[e] * (1.f / 6.f);
    float x2 = x * x, x6 = x2 * x2 * x2;
    float f = 1.f - 28.f * x6 + 48.f * x6 * x - 21.f * x6 * x2;
    sCut[tid] = (x < 1.f) ? f : 0.f;
  }
  for (int i = tid; i < 16 * 160; i += 128) {          // gather lat_in (pad 136->160)
    int e = i / 160, c = i - e * 160;
    size_t eg = e0 + e;
    float v = 0.f;
    if (c < 64)       v = ninv[(size_t)eidx[eg] * 64 + c];
    else if (c < 128) v = ninv[(size_t)eidx[E_TOT + eg] * 64 + (c - 64)];
    else if (c < 136) v = einv[eg * 8 + (c - 128)];
    sA[i] = (_Float16)v;
  }
  for (int i = tid; i < 16 * 9; i += 128) sSh[i] = eattr[e0 * 9 + i];
  __syncthreads();

  for (int t = wv; t < 16; t += 4) {                   // GEMM1: K=160
    v8f acc = {};
    acc = wmma_rowtile<5>(sA, 160, W2aT, 160, t * 16, lane, acc);
    store_silu_h(acc, sB1, 256, t * 16, lane);
  }
  __syncthreads();
  for (int t = wv; t < 16; t += 4) {                   // GEMM2: K=256, *cutoff
    v8f acc = {};
    acc = wmma_rowtile<8>(sB1, 256, W2bT, 256, t * 16, lane, acc);
    const int n = lane & 15, hi = lane >> 4;
#pragma unroll
    for (int r = 0; r < 8; ++r) {
      int m = r + 8 * hi;
      float s = silu_f(acc[r]) * sCut[m];
      latents[(e0 + m) * 256 + t * 16 + n] = s;
      sB2[m * 256 + t * 16 + n] = (_Float16)s;
    }
  }
  __syncthreads();
  for (int t = wv; t < 12; t += 4) {                   // GEMM3: Wenv0, N=192
    v8f acc = {};
    acc = wmma_rowtile<8>(sB2, 256, Wenv0T, 256, t * 16, lane, acc);
    const int n = lane & 15, hi = lane >> 4;
#pragma unroll
    for (int r = 0; r < 8; ++r) sWall[(r + 8 * hi) * 192 + t * 16 + n] = acc[r];
  }
  __syncthreads();

  for (int i = tid; i < 16 * 96; i += 128)             // w0 = w_all[:,0]
    w0g[(e0 + i / 96) * 96 + (i % 96)] = sWall[(i / 96) * 192 + (i % 96)];
  if (tid < 16) cutg[e0 + tid] = sCut[tid];
  for (int i = tid; i < 16 * 96; i += 128) {           // env scatter (w_all[:,1])
    int e = i / 96, kc = i - e * 96, k = kc / 3, l = kc - k * 3;
    float w = sWall[e * 192 + 96 + kc] * sCut[e];
    float* dst = env0 + (size_t)sCtr[e] * 288 + k * 9;
    int off = LOFF_C[l], d = LDIM_C[l];
    for (int m = 0; m < d; ++m)
      atomic_add_g(dst + off + m, sSh[e * 9 + off + m] * w);
  }
}

// ---------------------------------------------------------------------------
// Node stage: env -> mix(Wenvlin) -> polynomial gate(Wprod) -> mix(Wfeat)
// one wave32 per node, lane = output channel j
// ---------------------------------------------------------------------------
__global__ void __launch_bounds__(128)
allegro_node_mix(const float* __restrict__ env_in, const float* __restrict__ ninv,
                 const float* __restrict__ Wenvlin, const float* __restrict__ Wprod,
                 const float* __restrict__ Wfeat, int lidx,
                 float* __restrict__ feat_out) {
  __shared__ float sE[4][32 * 9];
  const int lane = threadIdx.x & 31, wv = threadIdx.x >> 5;
  const int n = blockIdx.x * 4 + wv;

  const float* src = env_in + (size_t)n * 288 + lane * 9;
#pragma unroll
  for (int m = 0; m < 9; ++m) sE[wv][lane * 9 + m] = src[m] * (1.f / 32.f);
  __syncthreads();

  const float* WL = Wenvlin + (size_t)lidx * 3 * 32 * 32;
  float mixv[9];
#pragma unroll
  for (int m = 0; m < 9; ++m) {
    int l = (m == 0) ? 0 : (m < 4) ? 1 : 2;
    const float* Wm = WL + l * 1024 + lane;
    float acc = 0.f;
    for (int k = 0; k < 32; ++k) acc += sE[wv][k * 9 + m] * Wm[k * 32];
    mixv[m] = acc;
  }
  float s = mixv[0];
  const float* WP = Wprod + (size_t)lidx * 3 * 64 * 32;
  const float* nv = ninv + (size_t)n * 64;
  float ew[3];
#pragma unroll
  for (int c = 0; c < 3; ++c) {
    const float* Wc = WP + c * 64 * 32 + lane;
    float acc = 0.f;
    for (int t = 0; t < 64; ++t) acc += nv[t] * Wc[t * 32];
    ew[c] = acc;
  }
  float g = ew[0] + ew[1] * s + ew[2] * s * s;
  __syncthreads();
#pragma unroll
  for (int m = 0; m < 9; ++m) sE[wv][lane * 9 + m] = mixv[m] * g;
  __syncthreads();
  const float* WF = Wfeat + (size_t)lidx * 3 * 32 * 32;
  float* dst = feat_out + (size_t)n * 288 + lane * 9;
#pragma unroll
  for (int m = 0; m < 9; ++m) {
    int l = (m == 0) ? 0 : (m < 4) ? 1 : 2;
    const float* Wm = WF + l * 1024 + lane;
    float acc = 0.f;
    for (int k = 0; k < 32; ++k) acc += sE[wv][k * 9 + m] * Wm[k * 32];
    dst[m] = acc;
  }
}

// ---------------------------------------------------------------------------
// Edge pass 2: norms -> latent MLP2 -> latent update -> Wenv1 -> env1 scatter
// ---------------------------------------------------------------------------
__global__ void __launch_bounds__(128)
allegro_edge_pass2(const float* __restrict__ eattr, const int* __restrict__ eidx,
                   const float* __restrict__ feat0,
                   const _Float16* __restrict__ Wl1aT, const _Float16* __restrict__ Wl1bT,
                   const _Float16* __restrict__ Wenv1T,
                   float* __restrict__ latents, const float* __restrict__ cutg,
                   float* __restrict__ env1) {
  __shared__ __align__(32) _Float16 sA[16 * 352];
  __shared__ __align__(32) _Float16 sB1[16 * 256];
  __shared__ __align__(32) _Float16 sB2[16 * 256];
  __shared__ float sWall[16 * 96];
  __shared__ float sSh[16 * 9];
  __shared__ float sCut[16];
  __shared__ int   sCtr[16];

  const int tid = threadIdx.x, lane = tid & 31, wv = tid >> 5;
  const size_t e0 = (size_t)blockIdx.x * 16;

  if (tid < 16) { sCut[tid] = cutg[e0 + tid]; sCtr[tid] = eidx[e0 + tid]; }
  for (int i = tid; i < 16 * 256; i += 128) {
    int e = i / 256, c = i - e * 256;
    sA[e * 352 + c] = (_Float16)latents[(e0 + e) * 256 + c];
  }
  for (int i = tid; i < 16 * 9; i += 128) sSh[i] = eattr[e0 * 9 + i];
  for (int i = tid; i < 512; i += 128) {               // l_norms -> cols 256..351
    int e = i / 32, k = i - e * 32;
    const float* f = feat0 + (size_t)eidx[e0 + e] * 288 + k * 9;
#pragma unroll
    for (int l = 0; l < 3; ++l) {
      float ss = 1e-8f;
      for (int m = 0; m < LDIM_C[l]; ++m) {
        float v = f[LOFF_C[l] + m];
        ss += v * v;
      }
      sA[e * 352 + 256 + l * 32 + k] = (_Float16)__builtin_amdgcn_sqrtf(ss);
    }
  }
  __syncthreads();

  for (int t = wv; t < 16; t += 4) {                   // Wl1a: K=352
    v8f acc = {};
    acc = wmma_rowtile<11>(sA, 352, Wl1aT, 352, t * 16, lane, acc);
    store_silu_h(acc, sB1, 256, t * 16, lane);
  }
  __syncthreads();
  for (int t = wv; t < 16; t += 4) {                   // Wl1b + latent update
    v8f acc = {};
    acc = wmma_rowtile<8>(sB1, 256, Wl1bT, 256, t * 16, lane, acc);
    const int n = lane & 15, hi = lane >> 4;
#pragma unroll
    for (int r = 0; r < 8; ++r) {
      int m = r + 8 * hi;
      float s = silu_f(acc[r]);
      size_t gi = (e0 + m) * 256 + t * 16 + n;
      float nl = C_OLD * latents[gi] + C_NEW * sCut[m] * s;
      latents[gi] = nl;
      sB2[m * 256 + t * 16 + n] = (_Float16)nl;
    }
  }
  __syncthreads();
  for (int t = wv; t < 6; t += 4) {                    // Wenv1: N=96
    v8f acc = {};
    acc = wmma_rowtile<8>(sB2, 256, Wenv1T, 256, t * 16, lane, acc);
    const int n = lane & 15, hi = lane >> 4;
#pragma unroll
    for (int r = 0; r < 8; ++r) sWall[(r + 8 * hi) * 96 + t * 16 + n] = acc[r];
  }
  __syncthreads();
  for (int i = tid; i < 16 * 96; i += 128) {           // env1 scatter
    int e = i / 96, kc = i - e * 96, k = kc / 3, l = kc - k * 3;
    float w = sWall[e * 96 + kc] * sCut[e];
    float* dst = env1 + (size_t)sCtr[e] * 288 + k * 9;
    int off = LOFF_C[l], d = LDIM_C[l];
    for (int m = 0; m < d; ++m)
      atomic_add_g(dst + off + m, sSh[e * 9 + off + m] * w);
  }
}

// ---------------------------------------------------------------------------
// Edge pass 3: Wfinlat GEMM fused into scal (no wfin materialization),
// then readout silu(ro@Wro1)@Wro2 -> out
// scal[e,l,j] = w0[e,j,l] * sum_k wfin[e,l,k,j] * P[e,k,l]
// ---------------------------------------------------------------------------
__global__ void __launch_bounds__(128)
allegro_edge_pass3(const float* __restrict__ eattr, const int* __restrict__ eidx,
                   const float* __restrict__ feat1, const float* __restrict__ w0g,
                   const float* __restrict__ latents,
                   const _Float16* __restrict__ WfinT, const _Float16* __restrict__ Wro1T,
                   const float* __restrict__ Wro2, float* __restrict__ out) {
  __shared__ __align__(32) _Float16 sLat[16 * 256];
  __shared__ __align__(32) _Float16 sRo[16 * 352];
  __shared__ __align__(32) _Float16 sAct[16 * 256];
  __shared__ float sP[16 * 32 * 3];
  __shared__ float sW0[16 * 96];
  __shared__ float sSh[16 * 9];
  __shared__ float sScal[16 * 96];
  __shared__ float sPart[128];

  const int tid = threadIdx.x, lane = tid & 31, wv = tid >> 5;
  const size_t e0 = (size_t)blockIdx.x * 16;

  for (int i = tid; i < 16 * 256; i += 128) {
    int e = i / 256, c = i - e * 256;
    float v = latents[(e0 + e) * 256 + c];
    sLat[i] = (_Float16)v;
    sRo[e * 352 + c] = (_Float16)v;
  }
  for (int i = tid; i < 16 * 96; i += 128) {
    sW0[i] = w0g[(e0 + i / 96) * 96 + (i % 96)];
    sScal[i] = 0.f;
  }
  for (int i = tid; i < 16 * 9; i += 128) sSh[i] = eattr[e0 * 9 + i];
  for (int i = tid; i < 512; i += 128) {               // P[e,k,l]
    int e = i / 32, k = i - e * 32;
    const float* f = feat1 + (size_t)eidx[e0 + e] * 288 + k * 9;
#pragma unroll
    for (int l = 0; l < 3; ++l) {
      float acc = 0.f;
      for (int m = 0; m < LDIM_C[l]; ++m)
        acc += f[LOFF_C[l] + m] * sSh[e * 9 + LOFF_C[l] + m];
      sP[(e * 32 + k) * 3 + l] = acc;
    }
  }
  __syncthreads();

  for (int t = wv; t < 192; t += 4) {                  // Wfinlat: 192 col tiles
    const int col0 = t * 16;
    v8f acc = {};
    acc = wmma_rowtile<8>(sLat, 256, WfinT, 256, col0, lane, acc);
    const int l = col0 >> 10, rem = col0 & 1023, k = rem >> 5, j0 = rem & 31;
    const int n = lane & 15, hi = lane >> 4;
    const int j = j0 + n;
#pragma unroll
    for (int r = 0; r < 8; ++r) {
      int e = r + 8 * hi;
      atomicAdd(&sScal[e * 96 + l * 32 + j], acc[r] * sP[(e * 32 + k) * 3 + l]);
    }
  }
  __syncthreads();
  for (int i = tid; i < 16 * 96; i += 128) {           // finalize scal -> ro cols
    int e = i / 96, c = i - e * 96, l = c / 32, j = c - l * 32;
    sRo[e * 352 + 256 + c] = (_Float16)(sScal[i] * sW0[e * 96 + j * 3 + l]);
  }
  __syncthreads();
  for (int t = wv; t < 16; t += 4) {                   // Wro1: K=352
    v8f acc = {};
    acc = wmma_rowtile<11>(sRo, 352, Wro1T, 352, t * 16, lane, acc);
    store_silu_h(acc, sAct, 256, t * 16, lane);
  }
  __syncthreads();
  {                                                    // Wro2 dot (256 -> 1)
    int e = tid >> 3, part = tid & 7;
    float acc = 0.f;
    for (int t = part * 32; t < part * 32 + 32; ++t)
      acc += (float)sAct[e * 256 + t] * Wro2[t];
    sPart[tid] = acc;
  }
  __syncthreads();
  if (tid < 16) {
    float acc = 0.f;
    for (int p = 0; p < 8; ++p) acc += sPart[tid * 8 + p];
    out[e0 + tid] = acc;
  }
}

// ---------------------------------------------------------------------------
extern "C" void kernel_launch(void* const* d_in, const int* in_sizes, int n_in,
                              void* d_out, int out_size, void* d_ws, size_t ws_size,
                              hipStream_t stream) {
  const float* eattr   = (const float*)d_in[0];
  const float* elen    = (const float*)d_in[1];
  const float* einv    = (const float*)d_in[2];
  const float* ninv    = (const float*)d_in[3];
  const float* W2a     = (const float*)d_in[4];
  const float* W2b     = (const float*)d_in[5];
  const float* Wl1a    = (const float*)d_in[6];
  const float* Wl1b    = (const float*)d_in[7];
  const float* Wenv0   = (const float*)d_in[8];
  const float* Wenv1   = (const float*)d_in[9];
  const float* Wenvlin = (const float*)d_in[10];
  const float* Wprod   = (const float*)d_in[11];
  const float* Wfeat   = (const float*)d_in[12];
  const float* Wfinlat = (const float*)d_in[13];
  const float* Wro1    = (const float*)d_in[14];
  const float* Wro2    = (const float*)d_in[15];
  const int*   eidx    = (const int*)d_in[16];

  char* ws = (char*)d_ws;
  size_t off = 0;
  auto alloc = [&](size_t bytes) {
    size_t o = off;
    off = (off + bytes + 255) & ~(size_t)255;
    return o;
  };
  _Float16* W2aT   = (_Float16*)(ws + alloc((size_t)256 * 160 * 2));
  _Float16* W2bT   = (_Float16*)(ws + alloc((size_t)256 * 256 * 2));
  _Float16* Wenv0T = (_Float16*)(ws + alloc((size_t)192 * 256 * 2));
  _Float16* Wl1aT  = (_Float16*)(ws + alloc((size_t)256 * 352 * 2));
  _Float16* Wl1bT  = (_Float16*)(ws + alloc((size_t)256 * 256 * 2));
  _Float16* Wenv1T = (_Float16*)(ws + alloc((size_t)96 * 256 * 2));
  _Float16* WfinT  = (_Float16*)(ws + alloc((size_t)3072 * 256 * 2));
  _Float16* Wro1T  = (_Float16*)(ws + alloc((size_t)256 * 352 * 2));
  float* latents = (float*)(ws + alloc((size_t)E_TOT * 256 * 4));
  float* cutg    = (float*)(ws + alloc((size_t)E_TOT * 4));
  float* w0g     = (float*)(ws + alloc((size_t)E_TOT * 96 * 4));
  float* env0    = (float*)(ws + alloc((size_t)NN_TOT * 288 * 4));
  float* feat0   = (float*)(ws + alloc((size_t)NN_TOT * 288 * 4));
  float* env1    = (float*)(ws + alloc((size_t)NN_TOT * 288 * 4));
  float* feat1   = (float*)(ws + alloc((size_t)NN_TOT * 288 * 4));

  hipMemsetAsync(env0, 0, (size_t)NN_TOT * 288 * 4, stream);
  hipMemsetAsync(env1, 0, (size_t)NN_TOT * 288 * 4, stream);

  // weight conversions (cheap, deterministic each call)
  allegro_prep_weight<<<512, 256, 0, stream>>>(W2a,   W2aT,   136, 256, 160);
  allegro_prep_weight<<<512, 256, 0, stream>>>(W2b,   W2bT,   256, 256, 256);
  allegro_prep_weight<<<512, 256, 0, stream>>>(Wenv0, Wenv0T, 256, 192, 256);
  allegro_prep_weight<<<512, 256, 0, stream>>>(Wl1a,  Wl1aT,  352, 256, 352);
  allegro_prep_weight<<<512, 256, 0, stream>>>(Wl1b,  Wl1bT,  256, 256, 256);
  allegro_prep_weight<<<512, 256, 0, stream>>>(Wenv1, Wenv1T, 256, 96,  256);
  allegro_prep_weight<<<2048, 256, 0, stream>>>(Wfinlat, WfinT, 256, 3072, 256);
  allegro_prep_weight<<<512, 256, 0, stream>>>(Wro1,  Wro1T,  352, 256, 352);

  const int EBLK = E_TOT / 16;     // 20000 tiles of 16 edges
  const int NBLK = NN_TOT / 4;     // one wave32 per node

  allegro_edge_pass1<<<EBLK, 128, 0, stream>>>(eattr, elen, einv, ninv, eidx,
                                               W2aT, W2bT, Wenv0T,
                                               latents, cutg, w0g, env0);
  allegro_node_mix<<<NBLK, 128, 0, stream>>>(env0, ninv, Wenvlin, Wprod, Wfeat, 0, feat0);
  allegro_edge_pass2<<<EBLK, 128, 0, stream>>>(eattr, eidx, feat0,
                                               Wl1aT, Wl1bT, Wenv1T,
                                               latents, cutg, env1);
  allegro_node_mix<<<NBLK, 128, 0, stream>>>(env1, ninv, Wenvlin, Wprod, Wfeat, 1, feat1);
  allegro_edge_pass3<<<EBLK, 128, 0, stream>>>(eattr, eidx, feat1, w0g, latents,
                                               WfinT, Wro1T, Wro2, (float*)d_out);
}
